// HetSAGPooling_28681791603116
// MI455X (gfx1250) — matrix operand order repaired and checked by reading, not compiled, not run on hardware
//
#include <hip/hip_runtime.h>
#include <stdint.h>

#define C_DIM 128
#define EDGE_UNROLL 4   // edges per thread in edge passes (b128 async stage = 4 ints/lane)

typedef __attribute__((ext_vector_type(16))) _Float16 v16h;
typedef __attribute__((ext_vector_type(8)))  float    v8f;

// ---- order-preserving float<->int maps (for atomic max / sort keys) ----
__device__ __forceinline__ int f32_ord_i(float f) {
  int i = __float_as_int(f);
  return (i < 0) ? (i ^ 0x7FFFFFFF) : i;          // monotonic, sign bit preserved (involution)
}
__device__ __forceinline__ float f32_unord_i(int i) {
  return __int_as_float((i < 0) ? (i ^ 0x7FFFFFFF) : i);
}
__device__ __forceinline__ unsigned f32_ord_u(float f) {
  unsigned u = __float_as_uint(f);
  return (u & 0x80000000u) ? ~u : (u | 0x80000000u);
}
__device__ __forceinline__ float f32_unord_u(unsigned u) {
  return __uint_as_float((u & 0x80000000u) ? (u & 0x7FFFFFFFu) : ~u);
}

// LDS byte offset of a __shared__ object (addrspace(3) pointers are offsets from LDS base)
__device__ __forceinline__ unsigned lds_off_u32(const void* p) {
  return (unsigned)(uintptr_t)p;
}

// ---------------------------------------------------------------------------
// Kernel 1: fused k/q/v projection via v_wmma_f32_16x16x32_f16.
// One wave = one 16-node tile; fully unrolled K loop -> 4 back-to-back WMMAs.
// B columns: 0=Wk, 1=Wq, 2=Wv, 3..15=0.
// A k order used here (contiguous, same as B; not the ISA table order; 16-bit A 16x32): lane l (m=l&15, hi=l>>4):
//   half j -> K = kb + hi*16 + j
// B layout (16-bit B 32x16): lane l (n=l&15, hi=l>>4): half j -> K = kb + hi*16 + j
// D layout (f32 C/D 16x16): lane l vgpr j -> M = j + hi*8, N = l&15
// ---------------------------------------------------------------------------
__global__ void __launch_bounds__(256)
proj_wmma(const float* __restrict__ x_inst, const float* __restrict__ x_net,
          const float* Wk_i, const float* bk_i, const float* Wq_i, const float* bq_i,
          const float* Wv_i, const float* bv_i,
          const float* Wk_n, const float* bk_n, const float* Wq_n, const float* bq_n,
          const float* Wv_n, const float* bv_n,
          float* __restrict__ k_all, float* __restrict__ q_all, float* __restrict__ v_all,
          int N_INST, int NT)
{
  const int tile   = (blockIdx.x * blockDim.x + threadIdx.x) >> 5;   // global wave id
  const int ntiles = NT >> 4;                                        // N_INST % 16 == 0
  if (tile >= ntiles) return;                                        // wave-uniform exit
  const int lane = threadIdx.x & 31;
  const int m    = lane & 15;
  const int hi   = lane >> 4;
  const int nodeBase = tile << 4;
  const bool isInst  = nodeBase < N_INST;

  const float* xrow = isInst ? (x_inst + (size_t)(nodeBase + m) * C_DIM)
                             : (x_net  + (size_t)(nodeBase - N_INST + m) * C_DIM);
  const float* wcol = isInst ? ((m == 1) ? Wq_i : (m == 2) ? Wv_i : Wk_i)
                             : ((m == 1) ? Wq_n : (m == 2) ? Wv_n : Wk_n);
  const float wmask = (m < 3) ? 1.0f : 0.0f;   // zero-fill unused B columns (no divergence)

  v8f acc = {};
#pragma unroll
  for (int kb = 0; kb < C_DIM; kb += 32) {
    v16h a, b;
#pragma unroll
    for (int j = 0; j < 8; ++j) {
      a[j]     = (_Float16)xrow[kb + hi * 16 + j];
      a[j + 8] = (_Float16)xrow[kb + hi * 16 + 8 + j];
    }
#pragma unroll
    for (int j = 0; j < 16; ++j)
      b[j] = (_Float16)(wcol[kb + hi * 16 + j] * wmask);
    // (neg_a, A, neg_b, B, c_mod, C, reuse_a, reuse_b)
    acc = __builtin_amdgcn_wmma_f32_16x16x32_f16(false, a, false, b, (short)0, acc,
                                                 false, false);
  }

  if (m < 3) {   // lanes N=0/1/2 (both hi halves) hold k/q/v results
    float bias = isInst ? ((m == 0) ? bk_i[0] : (m == 1) ? bq_i[0] : bv_i[0])
                        : ((m == 0) ? bk_n[0] : (m == 1) ? bq_n[0] : bv_n[0]);
    float* dst = (m == 0) ? k_all : (m == 1) ? q_all : v_all;
#pragma unroll
    for (int j = 0; j < 8; ++j)
      dst[nodeBase + hi * 8 + j] = acc[j] + bias;   // global ids: inst [0,Ni), net [Ni,NT)
  }
}

// ---------------------------------------------------------------------------
// Workspace init
// ---------------------------------------------------------------------------
__global__ void init_node_ws(int* mx_ord, float* den, float* num, int NT)
{
  int n = blockIdx.x * blockDim.x + threadIdx.x;
  if (n >= NT) return;
  mx_ord[n] = (int)0x80000000;   // < ordered(x) for every real float
  den[n] = 0.0f;
  num[n] = 0.0f;
}

// ---------------------------------------------------------------------------
// Edge pass 1: per-dst segment max of alpha.
// Index streams staged via global_load_async_to_lds_b128 (ASYNCcnt path):
// each lane pulls 16B of the src row and 16B of the dst row into LDS, then
// processes 4 edges from LDS. Node arrays are L2-resident gathers.
// ---------------------------------------------------------------------------
__global__ void __launch_bounds__(256)
edge_max(const int* __restrict__ ei, int E,
         const float* __restrict__ q_all, const float* __restrict__ k_all,
         int src_off, int dst_off,
         const float* arel, const float* prel,
         int* __restrict__ mx_ord)
{
  __shared__ int ssrc[256 * EDGE_UNROLL];
  __shared__ int sdst[256 * EDGE_UNROLL];
  const int t  = blockIdx.x * blockDim.x + threadIdx.x;
  const int e0 = t * EDGE_UNROLL;
  if (e0 >= E) return;                       // E % (256*4) == 0 -> no partial waves
  if (e0 + 32768 < 2 * E) __builtin_prefetch(ei + e0 + 32768, 0, 0);  // global_prefetch_b8

  const unsigned so = lds_off_u32(&ssrc[threadIdx.x * EDGE_UNROLL]);
  const unsigned dofs = lds_off_u32(&sdst[threadIdx.x * EDGE_UNROLL]);
  const int* gs = ei + e0;                   // 16B aligned (e0 % 4 == 0)
  const int* gd = ei + E + e0;               // E % 4 == 0
  asm volatile("global_load_async_to_lds_b128 %0, %1, off" :: "v"(so),   "v"(gs) : "memory");
  asm volatile("global_load_async_to_lds_b128 %0, %1, off" :: "v"(dofs), "v"(gd) : "memory");
  asm volatile("s_wait_asynccnt 0" ::: "memory");

  const float scale = arel[0] * prel[0];
#pragma unroll
  for (int u = 0; u < EDGE_UNROLL; ++u) {
    int e = e0 + u;
    if (e >= E) break;
    int s = ssrc[threadIdx.x * EDGE_UNROLL + u];
    int d = sdst[threadIdx.x * EDGE_UNROLL + u];
    float a = q_all[dst_off + d] * k_all[src_off + s] * scale;
    atomicMax(&mx_ord[dst_off + d], f32_ord_i(a));
  }
}

__global__ void finalize_mx(const int* __restrict__ mx_ord, float* __restrict__ mx_f, int NT)
{
  int n = blockIdx.x * blockDim.x + threadIdx.x;
  if (n >= NT) return;
  int v = mx_ord[n];
  mx_f[n] = (v == (int)0x80000000) ? 0.0f : f32_unord_i(v);   // no-in-edge nodes -> 0
}

// ---------------------------------------------------------------------------
// Edge pass 2: den += exp(a-mx[dst]); num += v[src]*mrel*exp(a-mx[dst])
// (agg = num/(den+1e-16) since den is per-dst constant). Same async staging.
// ---------------------------------------------------------------------------
__global__ void __launch_bounds__(256)
edge_sum(const int* __restrict__ ei, int E,
         const float* __restrict__ q_all, const float* __restrict__ k_all,
         const float* __restrict__ v_all,
         int src_off, int dst_off,
         const float* arel, const float* prel, const float* mrel,
         const float* __restrict__ mx_f,
         float* __restrict__ den, float* __restrict__ num)
{
  __shared__ int ssrc[256 * EDGE_UNROLL];
  __shared__ int sdst[256 * EDGE_UNROLL];
  const int t  = blockIdx.x * blockDim.x + threadIdx.x;
  const int e0 = t * EDGE_UNROLL;
  if (e0 >= E) return;
  if (e0 + 32768 < 2 * E) __builtin_prefetch(ei + e0 + 32768, 0, 0);

  const unsigned so = lds_off_u32(&ssrc[threadIdx.x * EDGE_UNROLL]);
  const unsigned dofs = lds_off_u32(&sdst[threadIdx.x * EDGE_UNROLL]);
  const int* gs = ei + e0;
  const int* gd = ei + E + e0;
  asm volatile("global_load_async_to_lds_b128 %0, %1, off" :: "v"(so),   "v"(gs) : "memory");
  asm volatile("global_load_async_to_lds_b128 %0, %1, off" :: "v"(dofs), "v"(gd) : "memory");
  asm volatile("s_wait_asynccnt 0" ::: "memory");

  const float scale = arel[0] * prel[0];
  const float mr = mrel[0];
#pragma unroll
  for (int u = 0; u < EDGE_UNROLL; ++u) {
    int e = e0 + u;
    if (e >= E) break;
    int s = ssrc[threadIdx.x * EDGE_UNROLL + u];
    int d = sdst[threadIdx.x * EDGE_UNROLL + u];
    int dg = dst_off + d;
    int sg = src_off + s;
    float a  = q_all[dg] * k_all[sg] * scale;
    float ex = expf(a - mx_f[dg]);
    atomicAdd(&den[dg], ex);
    atomicAdd(&num[dg], v_all[sg] * mr * ex);
  }
}

// ---------------------------------------------------------------------------
// Per-node epilogue + sort-key construction.
// key = ordered(score)<<32 | (0xFFFFFFFF - n): descending sort reproduces
// top_k order with lowest-index-first ties. Pad entries get key 0.
// ---------------------------------------------------------------------------
__global__ void node_score(const float* __restrict__ den, const float* __restrict__ num,
                           const float* Wa_i, const float* ba_i,
                           const float* Wa_n, const float* ba_n,
                           const float* W_fc, const float* b_fc, const float* w_sel,
                           unsigned long long* __restrict__ keys,
                           int N_INST, int NT, int PAD)
{
  int n = blockIdx.x * blockDim.x + threadIdx.x;
  if (n >= PAD) return;
  if (n >= NT) { keys[n] = 0ull; return; }
  float agg = num[n] / (den[n] + 1e-16f);
  // jax.nn.gelu approximate=True (tanh form)
  float g = 0.5f * agg * (1.0f + tanhf(0.7978845608028654f * (agg + 0.044715f * agg * agg * agg)));
  bool isInst = n < N_INST;
  float attn = g * (isInst ? Wa_i[0] : Wa_n[0]) + (isInst ? ba_i[0] : ba_n[0]);
  attn = attn * W_fc[0] + b_fc[0];
  float ws = w_sel[0];
  float sc = tanhf(attn * ws / fabsf(ws));      // norm(w_sel) = |w_sel| for 1-element vector
  keys[n] = ((unsigned long long)f32_ord_u(sc) << 32)
          | (unsigned long long)(0xFFFFFFFFu - (unsigned)n);
}

// ---------------------------------------------------------------------------
// Descending bitonic sort step over 64-bit keys (PAD = power of two)
// ---------------------------------------------------------------------------
__global__ void bitonic_step(unsigned long long* __restrict__ keys, int j, int k)
{
  int i = blockIdx.x * blockDim.x + threadIdx.x;
  int ixj = i ^ j;
  if (ixj > i) {
    unsigned long long a = keys[i];
    unsigned long long b = keys[ixj];
    bool descBlock = ((i & k) == 0);
    if (descBlock ? (a < b) : (a > b)) { keys[i] = b; keys[ixj] = a; }
  }
}

// ---------------------------------------------------------------------------
// Output gather: out[row] = x_all[idx] * sc  (float4 vectorized, 32 quads/row)
// ---------------------------------------------------------------------------
__global__ void __launch_bounds__(256)
gather_out(const unsigned long long* __restrict__ keys,
           const float* __restrict__ x_inst, const float* __restrict__ x_net,
           float* __restrict__ out, int N_INST, int K)
{
  long long t = (long long)blockIdx.x * blockDim.x + threadIdx.x;
  if (t >= (long long)K * 32) return;
  int row = (int)(t >> 5);
  int qd  = (int)(t & 31);
  unsigned long long key = keys[row];
  unsigned idx = 0xFFFFFFFFu - (unsigned)(key & 0xFFFFFFFFull);
  float sc = f32_unord_u((unsigned)(key >> 32));
  const float4* src = (idx < (unsigned)N_INST)
      ? ((const float4*)x_inst + (size_t)idx * 32)
      : ((const float4*)x_net  + ((size_t)idx - (size_t)N_INST) * 32);
  float4 v = src[qd];
  float4* o = (float4*)out + (size_t)row * 32;
  o[qd] = make_float4(v.x * sc, v.y * sc, v.z * sc, v.w * sc);
}

// ---------------------------------------------------------------------------
extern "C" void kernel_launch(void* const* d_in, const int* in_sizes, int n_in,
                              void* d_out, int out_size, void* d_ws, size_t ws_size,
                              hipStream_t stream)
{
  (void)n_in; (void)ws_size;
  const float* x_inst = (const float*)d_in[0];
  const float* x_net  = (const float*)d_in[1];
  const int*   ei_i2n = (const int*)d_in[2];
  const int*   ei_n2i = (const int*)d_in[3];
  const float* Wk_i = (const float*)d_in[4];  const float* bk_i = (const float*)d_in[5];
  const float* Wq_i = (const float*)d_in[6];  const float* bq_i = (const float*)d_in[7];
  const float* Wv_i = (const float*)d_in[8];  const float* bv_i = (const float*)d_in[9];
  const float* Wa_i = (const float*)d_in[10]; const float* ba_i = (const float*)d_in[11];
  const float* Wk_n = (const float*)d_in[12]; const float* bk_n = (const float*)d_in[13];
  const float* Wq_n = (const float*)d_in[14]; const float* bq_n = (const float*)d_in[15];
  const float* Wv_n = (const float*)d_in[16]; const float* bv_n = (const float*)d_in[17];
  const float* Wa_n = (const float*)d_in[18]; const float* ba_n = (const float*)d_in[19];
  const float* arel_i2n = (const float*)d_in[20];
  const float* mrel_i2n = (const float*)d_in[21];
  const float* prel_i2n = (const float*)d_in[22];
  const float* arel_n2i = (const float*)d_in[23];
  const float* mrel_n2i = (const float*)d_in[24];
  const float* prel_n2i = (const float*)d_in[25];
  const float* W_fc  = (const float*)d_in[26];
  const float* b_fc  = (const float*)d_in[27];
  const float* w_sel = (const float*)d_in[28];
  float* out = (float*)d_out;

  const int N_INST = in_sizes[0] / C_DIM;   // 100000 (divisible by 16)
  const int N_NET  = in_sizes[1] / C_DIM;
  const int NT     = N_INST + N_NET;        // 200000
  const int E1     = in_sizes[2] / 2;       // 3.2M
  const int E2     = in_sizes[3] / 2;
  const int K      = out_size / C_DIM;      // 100000
  int PAD = 1; while (PAD < NT) PAD <<= 1;  // 262144

  // ---- workspace carve (~7.7 MB) ----
  char* w = (char*)d_ws;
  float* k_all  = (float*)w; w += (size_t)NT * 4;
  float* q_all  = (float*)w; w += (size_t)NT * 4;
  float* v_all  = (float*)w; w += (size_t)NT * 4;
  int*   mx_ord = (int*)w;   w += (size_t)NT * 4;
  float* mx_f   = (float*)w; w += (size_t)NT * 4;
  float* den    = (float*)w; w += (size_t)NT * 4;
  float* num    = (float*)w; w += (size_t)NT * 4;
  w = (char*)(((uintptr_t)w + 7) & ~(uintptr_t)7);
  unsigned long long* keys = (unsigned long long*)w;

  dim3 blk(256);
  init_node_ws<<<(NT + 255) / 256, blk, 0, stream>>>(mx_ord, den, num, NT);

  const int ntiles = NT / 16;   // one wave per 16-node tile
  proj_wmma<<<(ntiles * 32 + 255) / 256, blk, 0, stream>>>(
      x_inst, x_net,
      Wk_i, bk_i, Wq_i, bq_i, Wv_i, bv_i,
      Wk_n, bk_n, Wq_n, bq_n, Wv_n, bv_n,
      k_all, q_all, v_all, N_INST, NT);

  // i2n: src=inst (off 0), dst=net (off N_INST) ; n2i: src=net, dst=inst
  const int eg1 = (E1 / EDGE_UNROLL + 255) / 256;
  const int eg2 = (E2 / EDGE_UNROLL + 255) / 256;
  edge_max<<<eg1, blk, 0, stream>>>(ei_i2n, E1, q_all, k_all, 0, N_INST,
                                    arel_i2n, prel_i2n, mx_ord);
  edge_max<<<eg2, blk, 0, stream>>>(ei_n2i, E2, q_all, k_all, N_INST, 0,
                                    arel_n2i, prel_n2i, mx_ord);
  finalize_mx<<<(NT + 255) / 256, blk, 0, stream>>>(mx_ord, mx_f, NT);
  edge_sum<<<eg1, blk, 0, stream>>>(ei_i2n, E1, q_all, k_all, v_all, 0, N_INST,
                                    arel_i2n, prel_i2n, mrel_i2n, mx_f, den, num);
  edge_sum<<<eg2, blk, 0, stream>>>(ei_n2i, E2, q_all, k_all, v_all, N_INST, 0,
                                    arel_n2i, prel_n2i, mrel_n2i, mx_f, den, num);

  node_score<<<(PAD + 255) / 256, blk, 0, stream>>>(den, num, Wa_i, ba_i, Wa_n, ba_n,
                                                    W_fc, b_fc, w_sel, keys, N_INST, NT, PAD);

  // full descending bitonic sort of 64-bit keys (171 steps for PAD=2^18)
  for (int kk = 2; kk <= PAD; kk <<= 1)
    for (int j = kk >> 1; j > 0; j >>= 1)
      bitonic_step<<<PAD / 256, blk, 0, stream>>>(keys, j, kk);

  gather_out<<<(int)(((long long)K * 32 + 255) / 256), blk, 0, stream>>>(
      keys, x_inst, x_net, out, N_INST, K);
}
